// PatchCoreViT_21981642621299
// MI455X (gfx1250) — matrix operand order repaired and checked by reading, not compiled, not run on hardware
//
#include <hip/hip_runtime.h>

typedef __attribute__((ext_vector_type(16))) __bf16 v16bf;
typedef __attribute__((ext_vector_type(8)))  __bf16 v8bf;
typedef __attribute__((ext_vector_type(8)))  float  v8f;
typedef __attribute__((ext_vector_type(4)))  int    v4i;

#define DIM 768
#define NPATCH 576
#define KSTEPS (DIM / 32)          // 24
#define SLAB_PITCH 80              // 64B bf16 K-slab + 16B pad (bank-conflict free)
#define SLAB_BYTES (NPATCH * SLAB_PITCH)
#define EPSF 1e-6f
#define NEGBIG -3.0e38f

// Order-preserving float <-> uint encoding so atomicMax(uint) works for all finite floats.
__device__ __forceinline__ unsigned encf(float f) {
    unsigned b = __float_as_uint(f);
    return (b & 0x80000000u) ? ~b : (b | 0x80000000u);
}
__device__ __forceinline__ float decf(unsigned e) {
    unsigned b = (e & 0x80000000u) ? (e & 0x7fffffffu) : ~e;
    return __uint_as_float(b);
}

// CDNA5 async global->LDS copy (no VGPR data path, tracked by ASYNCcnt).
__device__ __forceinline__ void stage_chunk_async(const void* gsrc, void* ldst) {
#if __has_builtin(__builtin_amdgcn_global_load_async_to_lds_b128)
    __builtin_amdgcn_global_load_async_to_lds_b128(
        (__attribute__((address_space(1))) v4i*)(size_t)gsrc,
        (__attribute__((address_space(3))) v4i*)(unsigned)(size_t)ldst,
        0, 0);
#else
    *(float4*)ldst = *(const float4*)gsrc;
#endif
}
__device__ __forceinline__ void wait_async_writes() {
#if __has_builtin(__builtin_amdgcn_global_load_async_to_lds_b128)
    asm volatile("s_wait_asynccnt 0x0" ::: "memory");
#endif
}

// Pin the WMMA/DS-read software pipeline (frags prefetched 3 tiles ahead).
__device__ __forceinline__ void sched_tile_pipeline() {
#if __has_builtin(__builtin_amdgcn_sched_group_barrier)
    // masks: 0x008 = MFMA/WMMA, 0x100 = DS read
    __builtin_amdgcn_sched_group_barrier(0x100, 6, 0);   // frags 0,1,2
    #pragma unroll
    for (int i = 0; i < 15; ++i) {
        __builtin_amdgcn_sched_group_barrier(0x008, 1, 0);   // WMMA t
        __builtin_amdgcn_sched_group_barrier(0x100, 2, 0);   // frag t+3
    }
    __builtin_amdgcn_sched_group_barrier(0x008, 3, 0);   // WMMA 15,16,17
#endif
}

// ---------------------------------------------------------------------------
// Kernel 1: L2-normalize patch features, write bf16 row-major [NPATCH][DIM].
__global__ void norm_patches_kernel(const float* __restrict__ f,
                                    __bf16* __restrict__ out) {
    const int row = blockIdx.x;
    const int tid = threadIdx.x;
    const float* src = f + (size_t)row * DIM;
    float x0 = src[tid], x1 = src[tid + 256], x2 = src[tid + 512];
    float s = x0 * x0 + x1 * x1 + x2 * x2;
    #pragma unroll
    for (int m = 16; m >= 1; m >>= 1) s += __shfl_xor(s, m, 32);
    __shared__ float wsum[8];
    __shared__ float sc_sh;
    if ((tid & 31) == 0) wsum[tid >> 5] = s;
    __syncthreads();
    if (tid == 0) {
        float t = 0.f;
        #pragma unroll
        for (int i = 0; i < 8; ++i) t += wsum[i];
        sc_sh = 1.0f / (sqrtf(t) + EPSF);
    }
    __syncthreads();
    const float sc = sc_sh;
    __bf16* dst = out + (size_t)row * DIM;
    dst[tid]       = (__bf16)(x0 * sc);
    dst[tid + 256] = (__bf16)(x1 * sc);
    dst[tid + 512] = (__bf16)(x2 * sc);
}

// ---------------------------------------------------------------------------
// Kernel 2: bf16 WMMA GEMM with fused memory-row norm (bank streamed from HBM
// exactly once; Σm² accumulated in f32, dot products rescaled by 1/(√Σm²+ε)
// in the epilogue). Block = 256 threads (8 waves): (wave&3) -> 16-row chunk
// of the block's 64 memory rows, (wave>>2) -> patch-tile group (18 tiles).
// Double-buffered LDS A-slab refilled by ASYNC global->LDS DMA; A fragments
// in a 4-buffer rotation at prefetch distance 3, pinned by sched barriers.
__global__ void __launch_bounds__(256, 1)
gemm_max_kernel(const float* __restrict__ mem,
                const __bf16* __restrict__ A,
                float* __restrict__ partial, int M) {
    constexpr int NT = 18;
    __shared__ __align__(16) unsigned char ldsA[2 * SLAB_BYTES];
    __shared__ unsigned ldsMax[NPATCH];

    const int tid  = threadIdx.x;
    const int wv   = tid >> 5;
    const int lane = tid & 31;
    const int rowBase = blockIdx.x * 64 + (wv & 3) * 16;
    const int tileOff = (wv >> 2) * NT;

    for (int p = tid; p < NPATCH; p += 256) ldsMax[p] = 0u;

    // B-matrix (K x 16 rows) lane addressing: lanes 0-15 hold K k0..k0+7, k0+16..k0+23 of
    // row (lane&15); lanes 16-31 hold K k0+8..k0+15, k0+24..k0+31 of the same rows.
    const int r  = rowBase + (lane & 15);
    const int rc = (r < M) ? r : (M - 1);
    const int kh = lane >> 4;
    const float* bsrc = mem + (size_t)rc * DIM + kh * 8;

    // Per-thread staging offsets: 2304 16B chunks, 9 per thread.
    int aoffB[9], loffB[9];
    #pragma unroll
    for (int i = 0; i < 9; ++i) {
        const int c = tid + i * 256;
        const int arow = c >> 2, part = c & 3;
        aoffB[i] = (arow * DIM + part * 8) * 2; // bytes into bf16 A
        loffB[i] = arow * SLAB_PITCH + part * 16;
    }

    // Preamble: stage slab 0 (k0 = 0) via async DMA; load B(k0 = 0).
    #pragma unroll
    for (int i = 0; i < 9; ++i)
        stage_chunk_async((const char*)A + aoffB[i], ldsA + loffB[i]);
    float4 cb0 = *(const float4*)(bsrc);
    float4 cb1 = *(const float4*)(bsrc + 4);
    float4 cb2 = *(const float4*)(bsrc + 16);
    float4 cb3 = *(const float4*)(bsrc + 20);
    wait_async_writes();
    __syncthreads();

    v8f acc[NT];
    #pragma unroll
    for (int t = 0; t < NT; ++t) acc[t] = v8f{};
    float ssq = 0.f;   // per-lane partial Σm² (this lane's half of the row)

    for (int kk = 0; kk < KSTEPS; ++kk) {
        const int k0 = kk * 32;
        const unsigned char* slab = ldsA + (kk & 1) * SLAB_BYTES;
        unsigned char* nslab = ldsA + ((kk + 1) & 1) * SLAB_BYTES;

        // Far prefetch of streaming B rows (2 k-steps ahead).
        if (kk + 2 < KSTEPS) __builtin_prefetch(bsrc + k0 + 64, 0, 3);

        // Refill the other slab with async DMA; nslab's readers finished at
        // the barrier that ended iteration kk-1, so writes are safe now and
        // have the whole compute phase to land.
        if (kk + 1 < KSTEPS) {
            const char* Ak = (const char*)A + (size_t)(k0 + 32) * 2;
            #pragma unroll
            for (int i = 0; i < 9; ++i)
                stage_chunk_async(Ak + aoffB[i], nslab + loffB[i]);
        }

        // Raw (unscaled) B chunk -> bf16 fragment; accumulate Σm² in f32.
        v16bf bfrag;
        bfrag[0]  = (__bf16)cb0.x; bfrag[1]  = (__bf16)cb0.y;
        bfrag[2]  = (__bf16)cb0.z; bfrag[3]  = (__bf16)cb0.w;
        bfrag[4]  = (__bf16)cb1.x; bfrag[5]  = (__bf16)cb1.y;
        bfrag[6]  = (__bf16)cb1.z; bfrag[7]  = (__bf16)cb1.w;
        bfrag[8]  = (__bf16)cb2.x; bfrag[9]  = (__bf16)cb2.y;
        bfrag[10] = (__bf16)cb2.z; bfrag[11] = (__bf16)cb2.w;
        bfrag[12] = (__bf16)cb3.x; bfrag[13] = (__bf16)cb3.y;
        bfrag[14] = (__bf16)cb3.z; bfrag[15] = (__bf16)cb3.w;
        ssq = fmaf(cb0.x, cb0.x, ssq); ssq = fmaf(cb0.y, cb0.y, ssq);
        ssq = fmaf(cb0.z, cb0.z, ssq); ssq = fmaf(cb0.w, cb0.w, ssq);
        ssq = fmaf(cb1.x, cb1.x, ssq); ssq = fmaf(cb1.y, cb1.y, ssq);
        ssq = fmaf(cb1.z, cb1.z, ssq); ssq = fmaf(cb1.w, cb1.w, ssq);
        ssq = fmaf(cb2.x, cb2.x, ssq); ssq = fmaf(cb2.y, cb2.y, ssq);
        ssq = fmaf(cb2.z, cb2.z, ssq); ssq = fmaf(cb2.w, cb2.w, ssq);
        ssq = fmaf(cb3.x, cb3.x, ssq); ssq = fmaf(cb3.y, cb3.y, ssq);
        ssq = fmaf(cb3.z, cb3.z, ssq); ssq = fmaf(cb3.w, cb3.w, ssq);

        // Next-iteration B loads (simple reload; prefetch keeps them in L0/L2).
        if (kk + 1 < KSTEPS) {
            const float* nb = bsrc + k0 + 32;
            cb0 = *(const float4*)(nb);
            cb1 = *(const float4*)(nb + 4);
            cb2 = *(const float4*)(nb + 16);
            cb3 = *(const float4*)(nb + 20);
        }

        // A fragments: tile stride 16*SLAB_PITCH = 1280B -> DS immediate offsets.
        const unsigned char* fbase =
            slab + tileOff * (16 * SLAB_PITCH) + (lane & 15) * SLAB_PITCH + kh * 16;
        v8bf lo0, hi0, lo1, hi1, lo2, hi2, lo3, hi3;
        lo0 = *(const v8bf*)(const void*)(fbase);
        hi0 = *(const v8bf*)(const void*)(fbase + 32);
        lo1 = *(const v8bf*)(const void*)(fbase + 1280);
        hi1 = *(const v8bf*)(const void*)(fbase + 1280 + 32);
        lo2 = *(const v8bf*)(const void*)(fbase + 2560);
        hi2 = *(const v8bf*)(const void*)(fbase + 2560 + 32);
        #pragma unroll
        for (int t = 0; t < NT; ++t) {
            // Prefetch fragment t+3 into buf[(t+3)&3]; distinct from the
            // buffers holding fragments t, t+1, t+2 -> no WAR with WMMA t.
            if (t + 3 < NT) {
                const unsigned char* ap = fbase + (t + 3) * 1280;
                switch ((t + 3) & 3) {
                case 0: lo0 = *(const v8bf*)(const void*)ap;
                        hi0 = *(const v8bf*)(const void*)(ap + 32); break;
                case 1: lo1 = *(const v8bf*)(const void*)ap;
                        hi1 = *(const v8bf*)(const void*)(ap + 32); break;
                case 2: lo2 = *(const v8bf*)(const void*)ap;
                        hi2 = *(const v8bf*)(const void*)(ap + 32); break;
                default:lo3 = *(const v8bf*)(const void*)ap;
                        hi3 = *(const v8bf*)(const void*)(ap + 32); break;
                }
            }
            v16bf afrag;
            switch (t & 3) {
            case 0: afrag = __builtin_shufflevector(lo0, hi0,
                0,1,2,3,4,5,6,7,8,9,10,11,12,13,14,15); break;
            case 1: afrag = __builtin_shufflevector(lo1, hi1,
                0,1,2,3,4,5,6,7,8,9,10,11,12,13,14,15); break;
            case 2: afrag = __builtin_shufflevector(lo2, hi2,
                0,1,2,3,4,5,6,7,8,9,10,11,12,13,14,15); break;
            default:afrag = __builtin_shufflevector(lo3, hi3,
                0,1,2,3,4,5,6,7,8,9,10,11,12,13,14,15); break;
            }
            acc[t] = __builtin_amdgcn_wmma_f32_16x16x32_bf16(
                false, afrag, false, bfrag, (short)0, acc[t], false, false);
        }
        sched_tile_pipeline();

        // Async slab refill must be LDS-visible before anyone reads it.
        if (kk + 1 < KSTEPS) wait_async_writes();
        __syncthreads();
    }

    // Row norm: lane L and L+16 hold complementary K-halves of the same row;
    // one xor-16 shuffle gives every lane its row's full Σm².
    const float ssqFull = ssq + __shfl_xor(ssq, 16, 32);
    const float inv = 1.0f / (sqrtf(ssqFull) + EPSF);

    // Epilogue: C/D layout: lane = memory row (N), VGPR v = patch M=v
    // (lanes 0-15) or v+8 (lanes 16-31). Max over rows = max within lane-half.
    const bool valid = (r < M);
    #pragma unroll
    for (int t = 0; t < NT; ++t) {
        #pragma unroll
        for (int v = 0; v < 8; ++v) {
            float val = valid ? (acc[t][v] * inv) : NEGBIG;
            val = fmaxf(val, __shfl_xor(val, 8, 32));
            val = fmaxf(val, __shfl_xor(val, 4, 32));
            val = fmaxf(val, __shfl_xor(val, 2, 32));
            val = fmaxf(val, __shfl_xor(val, 1, 32));
            if ((lane & 15) == 0) {
                const int p = (tileOff + t) * 16 + v + (kh ? 8 : 0);
                atomicMax(&ldsMax[p], encf(val));
            }
        }
    }
    __syncthreads();
    for (int p = tid; p < NPATCH; p += 256)
        partial[(size_t)blockIdx.x * NPATCH + p] = decf(ldsMax[p]);
}

// ---------------------------------------------------------------------------
// Kernel 3: reduce per-block partial maxima; patch_score = 1 - max_sim.
__global__ void reduce_max_kernel(const float* __restrict__ partial,
                                  float* __restrict__ out, int nChunks) {
    const int p = blockIdx.x;
    float m = NEGBIG;
    for (int c = threadIdx.x; c < nChunks; c += 256)
        m = fmaxf(m, partial[(size_t)c * NPATCH + p]);
    #pragma unroll
    for (int s = 16; s >= 1; s >>= 1) m = fmaxf(m, __shfl_xor(m, s, 32));
    __shared__ float w[8];
    if ((threadIdx.x & 31) == 0) w[threadIdx.x >> 5] = m;
    __syncthreads();
    if (threadIdx.x == 0) {
        float mm = w[0];
        #pragma unroll
        for (int i = 1; i < 8; ++i) mm = fmaxf(mm, w[i]);
        out[p] = 1.0f - mm;
    }
}

// ---------------------------------------------------------------------------
// Kernel 4: image score = mean of top-k patch scores (576 elems -> trivial).
__global__ void topk_mean_kernel(float* __restrict__ out,
                                 const int* __restrict__ topk) {
    __shared__ float s[NPATCH];
    for (int p = threadIdx.x; p < NPATCH; p += 256) s[p] = out[p];
    __syncthreads();
    if (threadIdx.x == 0) {
        int k = *topk;
        if (k < 1) k = 1;
        if (k > NPATCH) k = NPATCH;
        float sum = 0.f;
        for (int i = 0; i < k; ++i) {
            float best = NEGBIG;
            int bi = 0;
            for (int p = 0; p < NPATCH; ++p)
                if (s[p] > best) { best = s[p]; bi = p; }
            sum += best;
            s[bi] = NEGBIG;
        }
        out[NPATCH] = sum / (float)k;
    }
}

// ---------------------------------------------------------------------------
extern "C" void kernel_launch(void* const* d_in, const int* in_sizes, int n_in,
                              void* d_out, int out_size, void* d_ws, size_t ws_size,
                              hipStream_t stream) {
    const float* f    = (const float*)d_in[0];   // [576, 768]
    const float* mem  = (const float*)d_in[1];   // [200000, 768]
    const int*   topk = (const int*)d_in[2];     // scalar
    float* out = (float*)d_out;                  // [576 scores] + [1 image score]

    const int M = in_sizes[1] / DIM;             // 200000

    // Workspace: bf16 A | per-block partial maxima (~8.1 MB)
    char* ws = (char*)d_ws;
    __bf16* Abf = (__bf16*)ws;
    size_t off = (size_t)NPATCH * DIM * sizeof(__bf16);
    off = (off + 255) & ~(size_t)255;
    float* partial = (float*)(ws + off);

    const int nChunks = (M + 63) / 64;           // 3125

    norm_patches_kernel<<<NPATCH, 256, 0, stream>>>(f, Abf);
    gemm_max_kernel<<<nChunks, 256, 0, stream>>>(mem, Abf, partial, M);
    reduce_max_kernel<<<NPATCH, 256, 0, stream>>>(partial, out, nChunks);
    topk_mean_kernel<<<1, 256, 0, stream>>>(out, topk);
}
